// SubgraphEncoder_49941879718343
// MI455X (gfx1250) — matrix operand order repaired and checked by reading, not compiled, not run on hardware
//
#include <hip/hip_runtime.h>
#include <hip/hip_bf16.h>

#define N_NODESC 100000
#define N_EDGESC 1600000
#define N_GRAPHSC 512
#define FD 128
#define BN_EPSC 1e-5f

typedef unsigned short u16;
typedef __attribute__((ext_vector_type(16))) __bf16 v16bf;
typedef __attribute__((ext_vector_type(8)))  float  v8f;

// ---------- helpers ----------
__device__ __forceinline__ u16 f2bf(float f) {
    unsigned int u = __float_as_uint(f);
    u += 0x7FFFu + ((u >> 16) & 1u);           // round-to-nearest-even
    return (u16)(u >> 16);
}

__device__ __forceinline__ v16bf load_frag(const u16* p0, const u16* p1) {
    union { v16bf v; uint4 q[2]; } u;
    u.q[0] = *(const uint4*)p0;                // 16B -> ds_load_b128 / global_load_b128
    u.q[1] = *(const uint4*)p1;
    return u.v;
}

// ---------- weight prep: f32 [k][n] -> bf16 transposed [n][k] ----------
__global__ void prep_weights(const float* W1a, const float* W1b,
                             const float* W2a, const float* W2b,
                             const float* W3a, const float* W3b, u16* out) {
    int gid = blockIdx.x * 256 + threadIdx.x;
    if (gid >= 6 * FD * FD) return;
    int m = gid / (FD * FD);
    int idx = gid - m * FD * FD;
    int n = idx / FD, k = idx - n * FD;
    const float* W = (m == 0) ? W1a : (m == 1) ? W1b : (m == 2) ? W2a
                   : (m == 3) ? W2b : (m == 4) ? W3a : W3b;
    out[m * FD * FD + n * FD + k] = f2bf(W[k * FD + n]);
}

// ---------- edge scatter: agg[dst] += x[src]  (one edge per wave) ----------
__global__ __launch_bounds__(256) void edge_scatter(const float* __restrict__ x,
                                                    const int* __restrict__ ei,
                                                    float* __restrict__ agg) {
    int e = blockIdx.x * 8 + (threadIdx.x >> 5);   // 8 waves/block, 1 edge/wave
    int c = threadIdx.x & 31;                      // 32 lanes * float4 = 128 feats
    if (e >= N_EDGESC) return;
    int src = ei[e];
    int dst = ei[N_EDGESC + e];
    const float4 v = *(const float4*)(x + (size_t)src * FD + c * 4);
    float* p = agg + (size_t)dst * FD + c * 4;
    unsafeAtomicAdd(p + 0, v.x);
    unsafeAtomicAdd(p + 1, v.y);
    unsafeAtomicAdd(p + 2, v.z);
    unsafeAtomicAdd(p + 3, v.w);
}

// ---------- fused GIN MLP: h = relu((x+agg)@Wa + ba)@Wb + bb ----------
// Block: 128 threads = 4 waves, each wave owns a 16x128 output stripe.
__global__ __launch_bounds__(128) void gin_mlp(const float* __restrict__ xin,
                                               const float* __restrict__ agg,
                                               const u16* __restrict__ Wat,  // bf16 [n][k]
                                               const float* __restrict__ ba,
                                               const u16* __restrict__ Wbt,  // bf16 [n][k]
                                               const float* __restrict__ bb,
                                               float* __restrict__ hout) {
    __shared__ __align__(16) u16 sX[64 * FD];      // 16 KB: x+agg tile, reused as mid tile
    const int tid = threadIdx.x;
    const int row_base = blockIdx.x * 64;

    // stage: sX = bf16(x + agg), 128-bit global loads, fully coalesced
    #pragma unroll
    for (int i = 0; i < 16; ++i) {
        int idx = tid + i * 128;
        int r = idx >> 5, c = idx & 31;
        int grow = row_base + r;
        if (grow < N_NODESC) {
            const float4 a = *(const float4*)(xin + (size_t)grow * FD + c * 4);
            const float4 b = *(const float4*)(agg + (size_t)grow * FD + c * 4);
            float sx = a.x + b.x, sy = a.y + b.y, sz = a.z + b.z, sw = a.w + b.w;
            unsigned int lo = (unsigned int)f2bf(sx) | ((unsigned int)f2bf(sy) << 16);
            unsigned int hi = (unsigned int)f2bf(sz) | ((unsigned int)f2bf(sw) << 16);
            *(uint2*)(sX + r * FD + c * 4) = make_uint2(lo, hi);
        }
    }
    __syncthreads();

    const int wave = tid >> 5, lane = tid & 31;
    const int wrow = row_base + wave * 16;
    if (wrow + 16 > N_NODESC) return;              // uniform per-wave exit

    const int m    = lane & 15;                    // A row / B-D column within tile
    const int half = lane >> 4;

    // A fragments (16x32 bf16 per K-step), per ISA layout:
    //  lanes 0-15: K = ks*32 + {0..7, 16..23}; lanes 16-31: +8
    const u16* xrow = sX + (wave * 16 + m) * FD;
    v16bf A[4];
    #pragma unroll
    for (int ks = 0; ks < 4; ++ks)
        A[ks] = load_frag(xrow + ks * 32 + 8 * half,
                          xrow + ks * 32 + 16 + 8 * half);

    u16* sMid = sX + wave * 16 * FD;               // overlay: wave-private region

    // ---- GEMM 1: mid = relu(A @ Wa + ba) ----
    #pragma unroll
    for (int t = 0; t < 8; ++t) {
        v8f acc = {};
        #pragma unroll
        for (int ks = 0; ks < 4; ++ks) {
            // B (32x16): K = ks*32 + 8*half + {0..7, 16..23}, N = 16t+m
            const u16* wp = Wat + (size_t)(16 * t + m) * FD + ks * 32 + 8 * half;
            v16bf B = load_frag(wp, wp + 16);
            acc = __builtin_amdgcn_wmma_f32_16x16x32_bf16(false, A[ks], false, B,
                                                          (short)0, acc, false, false);
        }
        float bias = ba[16 * t + m];
        #pragma unroll
        for (int r = 0; r < 8; ++r) {
            float v = fmaxf(acc[r] + bias, 0.0f);  // row = r + 8*half, col = 16t+m
            sMid[(r + 8 * half) * FD + 16 * t + m] = f2bf(v);
        }
    }
    asm volatile("" ::: "memory");                 // LDS is in-order within a wave

    // A2 fragments from mid tile
    const u16* mrow = sMid + m * FD;
    v16bf A2[4];
    #pragma unroll
    for (int ks = 0; ks < 4; ++ks)
        A2[ks] = load_frag(mrow + ks * 32 + 8 * half,
                           mrow + ks * 32 + 16 + 8 * half);

    // ---- GEMM 2: h = A2 @ Wb + bb ----
    #pragma unroll
    for (int t = 0; t < 8; ++t) {
        v8f acc = {};
        #pragma unroll
        for (int ks = 0; ks < 4; ++ks) {
            const u16* wp = Wbt + (size_t)(16 * t + m) * FD + ks * 32 + 8 * half;
            v16bf B = load_frag(wp, wp + 16);
            acc = __builtin_amdgcn_wmma_f32_16x16x32_bf16(false, A2[ks], false, B,
                                                          (short)0, acc, false, false);
        }
        float bias = bb[16 * t + m];
        #pragma unroll
        for (int r = 0; r < 8; ++r)
            hout[(size_t)(wrow + r + 8 * half) * FD + 16 * t + m] = acc[r] + bias;
    }
}

// ---------- BatchNorm stats: per-feature sum / sumsq ----------
__global__ __launch_bounds__(256) void bn_stats(const float* __restrict__ h,
                                                float* __restrict__ gsum,
                                                float* __restrict__ gsq) {
    __shared__ float ss[256], sq[256];
    int tid = threadIdx.x, f = tid & 127, p = tid >> 7;
    int rend = min(N_NODESC, (int)(blockIdx.x + 1) * 256);
    float s = 0.f, q = 0.f;
    for (int r = blockIdx.x * 256 + p; r < rend; r += 2) {
        float v = h[(size_t)r * FD + f];
        s += v; q += v * v;
    }
    ss[tid] = s; sq[tid] = q;
    __syncthreads();
    if (tid < 128) {
        unsafeAtomicAdd(&gsum[f], ss[tid] + ss[tid + 128]);
        unsafeAtomicAdd(&gsq[f],  sq[tid] + sq[tid + 128]);
    }
}

__global__ void bn_finalize(const float* gsum, const float* gsq,
                            float* mean, float* inv) {
    int f = threadIdx.x;
    if (f < FD) {
        float mu = gsum[f] / (float)N_NODESC;
        float var = gsq[f] / (float)N_NODESC - mu * mu;
        mean[f] = mu;
        inv[f] = rsqrtf(var + BN_EPSC);
    }
}

__global__ __launch_bounds__(256) void bn_apply(const float* __restrict__ h,
                                                const float* __restrict__ mean,
                                                const float* __restrict__ inv,
                                                const float* __restrict__ g,
                                                const float* __restrict__ be,
                                                float* __restrict__ out) {
    size_t gid = (size_t)blockIdx.x * 256 + threadIdx.x;   // one float4 per thread
    size_t row = gid >> 5;
    int c = (int)(gid & 31);
    if (row >= N_NODESC) return;
    const float4 v = *(const float4*)(h + row * FD + c * 4);
    float4 o;
    int f = c * 4;
    o.x = fmaxf((v.x - mean[f + 0]) * inv[f + 0] * g[f + 0] + be[f + 0], 0.f);
    o.y = fmaxf((v.y - mean[f + 1]) * inv[f + 1] * g[f + 1] + be[f + 1], 0.f);
    o.z = fmaxf((v.z - mean[f + 2]) * inv[f + 2] * g[f + 2] + be[f + 2], 0.f);
    o.w = fmaxf((v.w - mean[f + 3]) * inv[f + 3] * g[f + 3] + be[f + 3], 0.f);
    *(float4*)(out + row * FD + c * 4) = o;
}

// ---------- pooling ----------
__global__ __launch_bounds__(256) void count_nodes(const int* __restrict__ batch,
                                                   float* __restrict__ cnt) {
    int gid = blockIdx.x * 256 + threadIdx.x;
    if (gid < N_NODESC) unsafeAtomicAdd(&cnt[batch[gid]], 1.0f);
}

__global__ void inv_count(const float* cnt, float* invc) {
    int g = blockIdx.x * 256 + threadIdx.x;
    if (g < N_GRAPHSC) invc[g] = 1.0f / fmaxf(cnt[g], 1.0f);
}

__global__ __launch_bounds__(256) void pool_sum(const float* __restrict__ h,
                                                const int* __restrict__ batch,
                                                float* __restrict__ pool) {
    size_t gid = (size_t)blockIdx.x * 256 + threadIdx.x;
    size_t row = gid >> 5;
    int c = (int)(gid & 31);
    if (row >= N_NODESC) return;
    int g = batch[row];                                    // uniform per wave
    const float4 v = *(const float4*)(h + row * FD + c * 4);
    float* p = pool + (size_t)g * FD + c * 4;
    unsafeAtomicAdd(p + 0, v.x);
    unsafeAtomicAdd(p + 1, v.y);
    unsafeAtomicAdd(p + 2, v.z);
    unsafeAtomicAdd(p + 3, v.w);
}

// ---------- final head: z = concat(pool)/cnt @ Wl + bl ; L2-normalize ----------
__global__ __launch_bounds__(128) void final_head(const float* __restrict__ pool, // [3][512][128]
                                                  const float* __restrict__ invc,
                                                  const float* __restrict__ Wl,   // [384][128]
                                                  const float* __restrict__ bl,
                                                  float* __restrict__ out) {
    __shared__ float red[128];
    int g = blockIdx.x, o = threadIdx.x;
    float ic = invc[g];
    float z = bl[o];
    for (int l = 0; l < 3; ++l) {
        const float* pb = pool + ((size_t)l * N_GRAPHSC + g) * FD;
        for (int f = 0; f < FD; ++f)
            z += pb[f] * ic * Wl[(size_t)(l * FD + f) * FD + o];
    }
    red[o] = z * z;
    __syncthreads();
    for (int s = 64; s > 0; s >>= 1) {
        if (o < s) red[o] += red[o + s];
        __syncthreads();
    }
    float denom = fmaxf(sqrtf(red[0]), 1e-12f);
    out[(size_t)g * FD + o] = z / denom;
}

// ---------- host ----------
extern "C" void kernel_launch(void* const* d_in, const int* in_sizes, int n_in,
                              void* d_out, int out_size, void* d_ws, size_t ws_size,
                              hipStream_t stream) {
    const float* x   = (const float*)d_in[0];
    const int*   ei  = (const int*)d_in[1];
    const int*   bat = (const int*)d_in[2];
    const float* W1a = (const float*)d_in[3];  const float* b1a = (const float*)d_in[4];
    const float* W1b = (const float*)d_in[5];  const float* b1b = (const float*)d_in[6];
    const float* g1  = (const float*)d_in[7];  const float* be1 = (const float*)d_in[8];
    const float* W2a = (const float*)d_in[9];  const float* b2a = (const float*)d_in[10];
    const float* W2b = (const float*)d_in[11]; const float* b2b = (const float*)d_in[12];
    const float* g2  = (const float*)d_in[13]; const float* be2 = (const float*)d_in[14];
    const float* W3a = (const float*)d_in[15]; const float* b3a = (const float*)d_in[16];
    const float* W3b = (const float*)d_in[17]; const float* b3b = (const float*)d_in[18];
    const float* g3  = (const float*)d_in[19]; const float* be3 = (const float*)d_in[20];
    const float* Wl  = (const float*)d_in[21]; const float* bl  = (const float*)d_in[22];

    const size_t NF = (size_t)N_NODESC * FD;
    float* ws   = (float*)d_ws;
    float* agg  = ws;
    float* hpre = agg  + NF;
    float* h1   = hpre + NF;
    float* h2   = h1   + NF;
    float* h3   = h2   + NF;
    u16*   wt   = (u16*)(h3 + NF);                 // 6 * 128*128 bf16
    float* aux  = (float*)(wt + 6 * FD * FD);
    float* gsum = aux;          float* gsq  = gsum + FD;
    float* mean = gsq + FD;     float* inv  = mean + FD;
    float* pool = inv + FD;                        // [3][512][128]
    float* cnt  = pool + 3 * (size_t)N_GRAPHSC * FD;
    float* invc = cnt + N_GRAPHSC;

    // one-time-per-call prep
    prep_weights<<<(6 * FD * FD + 255) / 256, 256, 0, stream>>>(W1a, W1b, W2a, W2b, W3a, W3b, wt);
    hipMemsetAsync(cnt, 0, N_GRAPHSC * sizeof(float), stream);
    hipMemsetAsync(pool, 0, 3 * (size_t)N_GRAPHSC * FD * sizeof(float), stream);
    count_nodes<<<(N_NODESC + 255) / 256, 256, 0, stream>>>(bat, cnt);
    inv_count<<<2, 256, 0, stream>>>(cnt, invc);

    const int scatter_blocks = N_EDGESC / 8;               // 200000
    const int gemm_blocks    = (N_NODESC + 63) / 64;       // 1563
    const int stats_blocks   = (N_NODESC + 255) / 256;     // 391
    const int elem_blocks    = (int)((NF / 4 + 255) / 256);// 12500

    auto layer = [&](const float* xin, const u16* Wat, const float* ba,
                     const u16* Wbt, const float* bb, const float* g,
                     const float* be, float* hpost, float* poolbuf) {
        hipMemsetAsync(agg, 0, NF * sizeof(float), stream);
        edge_scatter<<<scatter_blocks, 256, 0, stream>>>(xin, ei, agg);
        gin_mlp<<<gemm_blocks, 128, 0, stream>>>(xin, agg, Wat, ba, Wbt, bb, hpre);
        hipMemsetAsync(gsum, 0, FD * sizeof(float), stream);
        hipMemsetAsync(gsq,  0, FD * sizeof(float), stream);
        bn_stats<<<stats_blocks, 256, 0, stream>>>(hpre, gsum, gsq);
        bn_finalize<<<1, 128, 0, stream>>>(gsum, gsq, mean, inv);
        bn_apply<<<elem_blocks, 256, 0, stream>>>(hpre, mean, inv, g, be, hpost);
        pool_sum<<<elem_blocks, 256, 0, stream>>>(hpost, bat, poolbuf);
    };

    layer(x,  wt + 0 * FD * FD, b1a, wt + 1 * FD * FD, b1b, g1, be1, h1, pool + 0 * (size_t)N_GRAPHSC * FD);
    layer(h1, wt + 2 * FD * FD, b2a, wt + 3 * FD * FD, b2b, g2, be2, h2, pool + 1 * (size_t)N_GRAPHSC * FD);
    layer(h2, wt + 4 * FD * FD, b3a, wt + 5 * FD * FD, b3b, g3, be3, h3, pool + 2 * (size_t)N_GRAPHSC * FD);

    final_head<<<N_GRAPHSC, 128, 0, stream>>>(pool, invc, Wl, bl, (float*)d_out);
}
